// RecRS_47158740910801
// MI455X (gfx1250) — matrix operand order repaired and checked by reading, not compiled, not run on hardware
//
#include <hip/hip_runtime.h>

// ---------------------------------------------------------------------------
// Leaky RNN on MI455X (gfx1250): persistent cooperative kernel, wave32 WMMA.
//   SEQ=512, BATCH=256, S=1024, O=256, alpha=0.1
//   128 WGs x 256 threads.  w_r/w_o slices pre-swizzled into LDS (B layout)
//   once; h tiles live in registers; g = tanh(h) exchanged via double-buffered
//   global bf16 buffer with one grid barrier per step.
//   Branch-free fast tanh (v_exp/v_rcp); explicit addrspace(1) pointers so
//   A/g/x/out traffic uses global_load/global_store; x_t loads hoisted above
//   the barrier; k-loops unrolled x8.  This revision: use clang ext_vector
//   u32x4 (not HIP's class-type uint4) for addrspace(1) 128-bit loads.
// ---------------------------------------------------------------------------

typedef __attribute__((ext_vector_type(16))) __bf16       v16bf;
typedef __attribute__((ext_vector_type(8)))  float        v8f;
typedef __attribute__((ext_vector_type(4)))  unsigned int u32x4;

#define AS1 __attribute__((address_space(1)))
typedef AS1 unsigned short gu16;
typedef AS1 const float    gcf32;
typedef AS1 float          gf32;

#define NUM_WG   128
#define TPB      256
#define SEQ      512
#define BATCH    256
#define SDIM     1024
#define ODIM     256
#define KT_COUNT 32          // 1024 / 32

__device__ __forceinline__ unsigned short f2bf(float f) {
  unsigned u = __builtin_bit_cast(unsigned, f);
  u = (u + 0x7FFFu + ((u >> 16) & 1u)) >> 16;       // round-to-nearest-even
  return (unsigned short)u;
}

// Branch-free tanh: copysign(1 - 2/(2^(2|x|*log2e) + 1), x).
// |x| large -> exp saturates to +inf -> rcp -> 0 -> result +/-1. No fixups.
__device__ __forceinline__ float fast_tanh(float x) {
  float ax = __builtin_fabsf(x);
  float e  = __builtin_amdgcn_exp2f(ax * 2.885390081777926815f); // 2*log2(e)
  float r  = __builtin_amdgcn_rcpf(e + 1.0f);
  float y  = 1.0f - 2.0f * r;
  return __builtin_copysignf(y, x);
}

// A-operand tile (16x32 bf16): lane holds row M=lane%16, K = base+{0..7} and
// base+16+{0..7}, base = 8*(lane>=16).  g is row-major [rows][1024] bf16.
__device__ __forceinline__ v16bf load_a(const gu16* g, int row, int kbase) {
  struct P { u32x4 lo, hi; } p;
  p.lo = *(const AS1 u32x4*)(g + (size_t)row * SDIM + kbase);
  p.hi = *(const AS1 u32x4*)(g + (size_t)row * SDIM + kbase + 16);
  return __builtin_bit_cast(v16bf, p);
}

__device__ __forceinline__ void grid_barrier(unsigned* cnt, unsigned* gen) {
  __syncthreads();
  if (threadIdx.x == 0) {
    __threadfence();   // make this WG's g stores visible device-wide
    unsigned my = __hip_atomic_load(gen, __ATOMIC_RELAXED, __HIP_MEMORY_SCOPE_AGENT);
    unsigned old = __hip_atomic_fetch_add(cnt, 1u, __ATOMIC_ACQ_REL, __HIP_MEMORY_SCOPE_AGENT);
    if (old == NUM_WG - 1) {
      __hip_atomic_store(cnt, 0u, __ATOMIC_RELAXED, __HIP_MEMORY_SCOPE_AGENT);
      __hip_atomic_store(gen, my + 1u, __ATOMIC_RELEASE, __HIP_MEMORY_SCOPE_AGENT);
    } else {
      while (__hip_atomic_load(gen, __ATOMIC_ACQUIRE, __HIP_MEMORY_SCOPE_AGENT) == my) {
        __builtin_amdgcn_s_sleep(8);
      }
    }
    __threadfence();
  }
  __syncthreads();
}

__global__ void rnn_init(const float* __restrict__ h_init,
                         unsigned short* __restrict__ g0,
                         unsigned* __restrict__ cnt, unsigned* __restrict__ gen) {
  int i = blockIdx.x * blockDim.x + threadIdx.x;
  if (i == 0) { *cnt = 0u; *gen = 0u; }
  if (i < BATCH * SDIM) g0[i] = f2bf(fast_tanh(h_init[i]));
}

__global__ __launch_bounds__(TPB, 1)
void rnn_main(const float* __restrict__ xg,    const float* __restrict__ h_init,
              const float* __restrict__ w_r,   const float* __restrict__ b_r,
              const float* __restrict__ w_o,   const float* __restrict__ b_o,
              float* __restrict__ outg,
              unsigned short* __restrict__ g0p, unsigned short* __restrict__ g1p,
              unsigned* __restrict__ cnt,      unsigned* __restrict__ gen) {
  // Pre-swizzled WMMA-B operand slabs: [kt][lane][16 bf16], 32 KB each.
  __shared__ __align__(32) unsigned short ldsBr[KT_COUNT * 32 * 16];
  __shared__ __align__(32) unsigned short ldsBo[KT_COUNT * 32 * 16];

  // Explicit global address-space views (forces global_load/global_store).
  gcf32* X   = (gcf32*)(unsigned long long)xg;
  gf32*  OUT = (gf32*) (unsigned long long)outg;
  gu16*  G0  = (gu16*) (unsigned long long)g0p;
  gu16*  G1  = (gu16*) (unsigned long long)g1p;

  const int b    = blockIdx.x;
  const int tid  = threadIdx.x;
  const int lane = tid & 31;
  const int wave = tid >> 5;
  const int n    = lane & 15;
  const int hihalf = (lane >> 4);          // 0 for lanes 0-15, 1 for 16-31

  const int colR = 16 * (b & 63);          // recurrent state-col slice origin
  const int bh   = b >> 6;                 // batch half (0/1)
  const int colO = 16 * (b >> 3);          // output-col slice origin

  // --- one-time weight preload into LDS, swizzled to B layout ------------
  // B layout (16-bit, 32x16): lane holds N=lane%16, K = 8*(lane/16)+{0..7} and 16+8*(lane/16)+{0..7}.
  for (int i = tid; i < KT_COUNT * 32 * 16; i += TPB) {
    int kt = i >> 9, rem = i & 511, ln = rem >> 4, j = rem & 15;
    int k  = 32 * kt + ((ln >= 16) ? 8 : 0) + j + ((j >= 8) ? 8 : 0);
    int nn = ln & 15;
    ldsBr[i] = f2bf(w_r[(size_t)(colR + nn) * SDIM + k]);
    ldsBo[i] = f2bf(w_o[(size_t)(colO + nn) * SDIM + k]);
  }
  __syncthreads();

  // --- persistent per-lane state (WMMA C/D layout) ------------------------
  // h tile: rows [r0, r0+16), cols [colR, colR+16).
  // reg r, lane l: row = r0 + r + 8*(l>=16), col = colR + l%16.
  const int r0   = 128 * bh + 16 * wave;
  const int colg = colR + n;
  const float br_bias = b_r[colg];
  float h[8];
#pragma unroll
  for (int r = 0; r < 8; ++r)
    h[r] = h_init[(size_t)(r0 + r + 8 * hihalf) * SDIM + colg];

  const bool do_err = (wave < 2);
  const int  erow0  = 16 * ((b & 7) * 2 + wave);      // m-tile origin (valid if do_err)
  const float bo_bias = b_o[colO + n];

  const v16bf* Brv = reinterpret_cast<const v16bf*>(ldsBr);
  const v16bf* Bov = reinterpret_cast<const v16bf*>(ldsBo);
  const int kb = 8 * hihalf;                           // A-operand K base offset

  for (int t = 0; t < SEQ; ++t) {
    gu16* gcur = (t & 1) ? G1 : G0;
    gu16* gnx  = (t & 1) ? G0 : G1;

    // ---- hoisted x_t loads (independent of the barrier; hide HBM latency) ----
    float xv[8];
    if (do_err) {
#pragma unroll
      for (int r = 0; r < 8; ++r) {
        size_t idx = ((size_t)t * BATCH + erow0 + r + 8 * hihalf) * ODIM + (colO + n);
        xv[r] = X[idx];
      }
    }

    // ---- recurrent GEMM: acc = tanh(h_t) @ w_r_slice ----
    v8f acc = {0.f, 0.f, 0.f, 0.f, 0.f, 0.f, 0.f, 0.f};
#pragma unroll 8
    for (int kt = 0; kt < KT_COUNT; ++kt) {
      v16bf a  = load_a(gcur, r0 + n, 32 * kt + kb);
      v16bf bm = Brv[kt * 32 + lane];
      acc = __builtin_amdgcn_wmma_f32_16x16x32_bf16(
                false, a, false, bm, (short)0, acc, false, false);
    }

    // ---- leaky update + tanh + publish g_{t+1} (bf16, global b16 stores) ----
#pragma unroll
    for (int r = 0; r < 8; ++r) {
      float hn = 0.9f * h[r] + 0.1f * (acc[r] + br_bias);
      h[r] = hn;
      gnx[(size_t)(r0 + r + 8 * hihalf) * SDIM + colg] = f2bf(fast_tanh(hn));
    }

    grid_barrier(cnt, gen);       // g_{t+1} now globally visible

    // ---- output GEMM: err_t = tanh(h_{t+1}) @ w_o_slice + b_o - x_t ----
    if (do_err) {
      v8f ao = {0.f, 0.f, 0.f, 0.f, 0.f, 0.f, 0.f, 0.f};
#pragma unroll 8
      for (int kt = 0; kt < KT_COUNT; ++kt) {
        v16bf a  = load_a(gnx, erow0 + n, 32 * kt + kb);
        v16bf bm = Bov[kt * 32 + lane];
        ao = __builtin_amdgcn_wmma_f32_16x16x32_bf16(
                 false, a, false, bm, (short)0, ao, false, false);
      }
#pragma unroll
      for (int r = 0; r < 8; ++r) {
        size_t idx = ((size_t)t * BATCH + erow0 + r + 8 * hihalf) * ODIM + (colO + n);
        OUT[idx] = ao[r] + bo_bias - xv[r];
      }
    }
  }
}

extern "C" void kernel_launch(void* const* d_in, const int* in_sizes, int n_in,
                              void* d_out, int out_size, void* d_ws, size_t ws_size,
                              hipStream_t stream) {
  const float* x      = (const float*)d_in[0];
  const float* h_init = (const float*)d_in[1];
  const float* w_r    = (const float*)d_in[2];
  const float* b_r    = (const float*)d_in[3];
  const float* w_o    = (const float*)d_in[4];
  const float* b_o    = (const float*)d_in[5];
  float* out = (float*)d_out;

  unsigned char* ws = (unsigned char*)d_ws;
  unsigned* cnt = (unsigned*)ws;
  unsigned* gen = cnt + 1;
  unsigned short* g0 = (unsigned short*)(ws + 256);
  unsigned short* g1 = g0 + (size_t)BATCH * SDIM;

  rnn_init<<<dim3((BATCH * SDIM + TPB - 1) / TPB), dim3(TPB), 0, stream>>>(
      h_init, g0, cnt, gen);
  rnn_main<<<dim3(NUM_WG), dim3(TPB), 0, stream>>>(
      x, h_init, w_r, b_r, w_o, b_o, out, g0, g1, cnt, gen);
}
